// GlobalDiscreteAttention_8160437863081
// MI455X (gfx1250) — matrix operand order repaired and checked by reading, not compiled, not run on hardware
//
#include <hip/hip_runtime.h>
#include <hip/hip_bf16.h>
#include <math.h>

typedef __attribute__((ext_vector_type(16))) _Float16 v16h;
typedef __attribute__((ext_vector_type(8)))  _Float16 v8h;
typedef __attribute__((ext_vector_type(8)))  float    v8f;

#define BATCH 4
#define KBITS 8
#define NPIX  4096   // 64*64
#define DDIM  64
#define NCODE 256
#define VTPAD (NCODE + 8)   // padded LDS row: 264 halves = 528 B (16B-aligned rows)

// -------- workspace layout --------
// [0, 16384)            uint8 codes[B][N]
// [16384, 20480)        int   hist[B][256]
// [20480, 282624)       float table[B][256][64]
#define WS_CODES 0
#define WS_HIST  (BATCH * NPIX)
#define WS_TABLE (WS_HIST + BATCH * NCODE * (int)sizeof(int))

// Kernel 1: per-pixel 8-bit code + per-batch histogram of evidence-key codes.
__global__ void k_codes_hist(const float* __restrict__ z,
                             const unsigned char* __restrict__ em,
                             unsigned char* __restrict__ codes,
                             int* __restrict__ hist) {
    __shared__ int hsh[NCODE];
    const int t = threadIdx.x;
    if (t < NCODE) hsh[t] = 0;
    __syncthreads();

    const int b = blockIdx.y;
    const int n = blockIdx.x * blockDim.x + t;
    const float* zb = z + (size_t)b * KBITS * NPIX + n;
    unsigned code = 0u;
#pragma unroll
    for (int k = 0; k < KBITS; ++k)
        code |= (zb[(size_t)k * NPIX] > 0.5f ? 1u : 0u) << k;
    codes[(size_t)b * NPIX + n] = (unsigned char)code;

    if (em[(size_t)b * NPIX + n])
        atomicAdd(&hsh[code], 1);
    __syncthreads();
    if (t < NCODE && hsh[t] != 0)
        atomicAdd(&hist[b * NCODE + t], hsh[t]);
}

// Kernel 2: per batch, build OUT_TABLE[256][64] via a 256x256x64 WMMA GEMM.
//   A[i][j] = exp(-popcount(i^j)/T) * cnt[j]   (f16, built in-register, reused over N)
//   B[j][d] = value_table[j][d]                (f16, LDS transposed -> ds_load_b128)
//   OUT[i][d] = (A @ B)[i][d] / max(Z[i], 1e-20),  Z[i] = sum_j A[i][j] (f32)
// 8 wave32s; wave w owns a 2x4 block of 16x16 tiles (rows 32w..32w+31, all 64 cols).
__global__ void k_table_wmma(const float* __restrict__ vt,
                             const float* __restrict__ temperature,
                             const int* __restrict__ hist,
                             float* __restrict__ table) {
    __shared__ _Float16 vt_t[DDIM][VTPAD];     // transposed value table, ~33 KB
    __shared__ float cnt_sh[NCODE];
    __shared__ float Z_sh[NCODE];
    __shared__ float ew_sh[KBITS + 1];

    const int b = blockIdx.x;
    const int t = threadIdx.x;   // 256 threads = 8 wave32s

    const float temp = fmaxf(temperature[0], 0.1f);
    if (t <= KBITS) ew_sh[t] = expf(-(float)t / temp);
    cnt_sh[t] = (float)hist[b * NCODE + t];
#pragma unroll
    for (int d = 0; d < DDIM; ++d)                       // transpose on stage-in
        vt_t[d][t] = (_Float16)vt[(size_t)t * DDIM + d]; // contiguous across t
    __syncthreads();

    // Row sums (softmax denominators) in f32.
    float zacc = 0.0f;
    for (int j = 0; j < NCODE; ++j)
        zacc += ew_sh[__popc(t ^ j)] * cnt_sh[j];
    Z_sh[t] = zacc;
    __syncthreads();

    const int wave = t >> 5;
    const int lane = t & 31;
    const int g    = lane >> 4;   // half-wave group (ISA 7.12.2 16-bit layouts)
    const int l15  = lane & 15;
    const int mt0  = wave * 2;    // this wave's first M tile

    v8f acc[2][4] = {};

    for (int ks = 0; ks < 8; ++ks) {    // K = 256 in steps of 32
        // ---- B fragments: one per N tile, 16 contiguous halves -> 2x b128 ----
        v16h bf[4];
#pragma unroll
        for (int nt = 0; nt < 4; ++nt) {
            const _Float16* p = &vt_t[nt * 16 + l15][ks * 32 + 16 * g];
            v8h lo = *(const v8h*)p;        // 16B-aligned (row 528B, off 64ks+32g)
            v8h hi = *(const v8h*)(p + 8);
            bf[nt] = __builtin_shufflevector(lo, hi,
                                             0, 1, 2, 3, 4, 5, 6, 7,
                                             8, 9, 10, 11, 12, 13, 14, 15);
        }
        // ---- A fragments: one per M tile, reused across all 4 N tiles ----
        v16h af[2];
#pragma unroll
        for (int m = 0; m < 2; ++m) {
            const int rowA = (mt0 + m) * 16 + l15;
#pragma unroll
            for (int r = 0; r < 8; ++r) {
                const int kb = ks * 32 + 16 * g + 2 * r;
                af[m][2 * r]     = (_Float16)(ew_sh[__popc(rowA ^ kb)]       * cnt_sh[kb]);
                af[m][2 * r + 1] = (_Float16)(ew_sh[__popc(rowA ^ (kb + 1))] * cnt_sh[kb + 1]);
            }
        }
        // ---- 2x4 register-blocked WMMA ----
#pragma unroll
        for (int m = 0; m < 2; ++m)
#pragma unroll
            for (int nt = 0; nt < 4; ++nt)
                acc[m][nt] = __builtin_amdgcn_wmma_f32_16x16x32_f16(
                    false, af[m], false, bf[nt], (short)0, acc[m][nt], false, false);
    }

#pragma unroll
    for (int m = 0; m < 2; ++m) {
#pragma unroll
        for (int r = 0; r < 8; ++r) {
            const int rowD = (mt0 + m) * 16 + r + 8 * g;  // C/D layout: VGPR r -> row
            const float inv_z = 1.0f / fmaxf(Z_sh[rowD], 1e-20f);
#pragma unroll
            for (int nt = 0; nt < 4; ++nt) {
                const int colD = nt * 16 + l15;
                table[((size_t)b * NCODE + rowD) * DDIM + colD] = acc[m][nt][r] * inv_z;
            }
        }
    }
}

// Kernel 3: scatter table rows into (B, D, H, W) output.
__global__ void k_scatter(const unsigned char* __restrict__ codes,
                          const float* __restrict__ table,
                          float* __restrict__ out) {
    const int b = blockIdx.y;
    const int n = blockIdx.x * blockDim.x + threadIdx.x;
    const int code = codes[(size_t)b * NPIX + n];
    const float* row = table + ((size_t)b * NCODE + code) * DDIM;
    float* ob = out + (size_t)b * DDIM * NPIX + n;
#pragma unroll
    for (int d = 0; d < DDIM; ++d)
        ob[(size_t)d * NPIX] = row[d];   // coalesced across n for each d
}

extern "C" void kernel_launch(void* const* d_in, const int* in_sizes, int n_in,
                              void* d_out, int out_size, void* d_ws, size_t ws_size,
                              hipStream_t stream) {
    (void)in_sizes; (void)n_in; (void)out_size; (void)ws_size;

    const float*         z    = (const float*)d_in[0];
    const unsigned char* em   = (const unsigned char*)d_in[1];  // jax bool: 1 byte
    const float*         temp = (const float*)d_in[2];
    const float*         vt   = (const float*)d_in[3];
    // d_in[4] mask_value: unused (masked keys carry exactly zero attention weight)
    // d_in[5] pop_lut:    unused (__popc)

    unsigned char* codes = (unsigned char*)d_ws + WS_CODES;
    int*           hist  = (int*)((char*)d_ws + WS_HIST);
    float*         table = (float*)((char*)d_ws + WS_TABLE);

    hipMemsetAsync(hist, 0, BATCH * NCODE * sizeof(int), stream);

    dim3 gridBN(NPIX / 256, BATCH);
    k_codes_hist<<<gridBN, 256, 0, stream>>>(z, em, codes, hist);
    k_table_wmma<<<BATCH, 256, 0, stream>>>(vt, temp, hist, table);
    k_scatter<<<gridBN, 256, 0, stream>>>(codes, table, (float*)d_out);
}
